// DirectMPIGO_79706003079271
// MI455X (gfx1250) — compile-verified, matrix operand order checked by reading, not run on hardware
//
#include <hip/hip_runtime.h>
#include <hip/hip_bf16.h>

typedef __attribute__((ext_vector_type(16))) _Float16 v16h;
typedef __attribute__((ext_vector_type(2)))  _Float16 v2h;
typedef __attribute__((ext_vector_type(8)))  float    v8f;

#define SDEPTH   128
#define HASH_CH  28
#define LAT_DIM  64
#define HID      128
#define OUT_F    28
#define WAVES    8
#define OSTRIDE  33   // padded sOut row stride (banks)

// ---------------------------------------------------------------------------
// Fragment-layout helper: 16-bit A-matrix 16x32 layout (CDNA5 ISA 7.12.2)
// lane = g*16 + M ; element i = 2*v+half holds K = (v<4?0:16) + g*8 + 2*(v&3)+half
// Inverse: given k_local (0..31):
//   v = (kl>>4)*4 + ((kl>>1)&3), g = (kl>>3)&1, half = kl&1
// LDS activation buffer: [wave][kt(4)][lane(32)][halves(16)]
// Note: for even k, frag_off(k)+1 == frag_off(k+1)  (only 'half' bit differs)
// ---------------------------------------------------------------------------
__device__ __forceinline__ int frag_off(int wv, int ml, int k) {
  int kt = k >> 5, kl = k & 31;
  int v  = ((kl >> 4) << 2) | ((kl >> 1) & 3);
  int g  = (kl >> 3) & 1;
  int hf = kl & 1;
  int ln = (g << 4) | ml;
  return ((wv * 4 + kt) * 32 + ln) * 16 + (v << 1) + hf;
}

// ---------------------------------------------------------------------------
// Tiny time-latent MLP: 1 -> 128 -> 128 -> 64, result stored as f16
// ---------------------------------------------------------------------------
__global__ __launch_bounds__(128) void latent_kernel(
    const float* __restrict__ times_t,
    const float* __restrict__ w1, const float* __restrict__ b1,
    const float* __restrict__ w2, const float* __restrict__ b2,
    const float* __restrict__ w3, const float* __restrict__ b3,
    _Float16* __restrict__ lat_out) {
  __shared__ float h1[HID], h2[HID];
  int j = threadIdx.x;
  float t0 = times_t[0];
  float a = t0 * w1[j] + b1[j];
  h1[j] = a > 0.f ? a : 0.f;
  __syncthreads();
  float s = b2[j];
  for (int k = 0; k < HID; ++k) s += h1[k] * w2[k * HID + j];
  h2[j] = s > 0.f ? s : 0.f;
  __syncthreads();
  if (j < LAT_DIM) {
    float s3 = b3[j];
    for (int k = 0; k < HID; ++k) s3 += h2[k] * w3[k * LAT_DIM + j];
    lat_out[j] = (_Float16)s3;
  }
}

// ---------------------------------------------------------------------------
// Repack MLP weights into WMMA B-fragment layout (f16), 1KB per fragment,
// lane-major so each lane's 16 halves are contiguous (b128-coalesced loads).
// Fragment order: L0: f = nt*3+kt (24) | L1: 24+nt*4+kt (32) |
//                 L2: 56+nt*4+kt (32) | L3: 88+nt*4+kt (8)
// ---------------------------------------------------------------------------
__global__ __launch_bounds__(32) void pack_kernel(
    const float* __restrict__ w0, const float* __restrict__ w1,
    const float* __restrict__ w2, const float* __restrict__ w3,
    _Float16* __restrict__ out) {
  int f = blockIdx.x, lane = threadIdx.x;
  const float* W; int kt, nt, Ksrc, Nsrc;
  if (f < 24)       { W = w0; kt = f % 3;        nt = f / 3;        Ksrc = HASH_CH + LAT_DIM; Nsrc = HID;   }
  else if (f < 56)  { int g = f - 24; W = w1; kt = g % 4; nt = g / 4; Ksrc = HID; Nsrc = HID;   }
  else if (f < 88)  { int g = f - 56; W = w2; kt = g % 4; nt = g / 4; Ksrc = HID; Nsrc = HID;   }
  else              { int g = f - 88; W = w3; kt = g % 4; nt = g / 4; Ksrc = HID; Nsrc = OUT_F; }
  int g16 = lane >> 4, nl = lane & 15;
  for (int i = 0; i < 16; ++i) {
    int v = i >> 1, hf = i & 1;
    int kl = ((v >> 2) << 4) | (g16 << 3) | ((v & 3) << 1) | hf;
    int k = kt * 32 + kl;
    int n = nt * 16 + nl;
    float val = (k < Ksrc && n < Nsrc) ? W[k * Nsrc + n] : 0.f;
    out[f * 512 + lane * 16 + i] = (_Float16)val;
  }
}

// ---------------------------------------------------------------------------
// One WMMA MLP layer for a wave's 16-sample tile.
// A frags preloaded to registers so in-place LDS writeback is safe (DS in-order).
// ---------------------------------------------------------------------------
template <int KTN, int NTN, bool FINAL>
__device__ __forceinline__ void mlp_layer(
    _Float16* sAct, float* sOut, const _Float16* __restrict__ wpack,
    int wave, int lane, int frag0, const float* __restrict__ bias, int nbias) {
  int actbase = wave * (4 * 32 * 16);
  v16h A[KTN];
#pragma unroll
  for (int kt = 0; kt < KTN; ++kt)
    A[kt] = *(const v16h*)&sAct[actbase + (kt * 32 + lane) * 16];

  int nl = lane & 15, g = lane >> 4;
#pragma unroll
  for (int nt = 0; nt < NTN; ++nt) {
    int n = nt * 16 + nl;
    float b0 = (n < nbias) ? bias[n] : 0.f;
    v8f acc;
#pragma unroll
    for (int i = 0; i < 8; ++i) acc[i] = b0;
#pragma unroll
    for (int kt = 0; kt < KTN; ++kt) {
      v16h Bf = *(const v16h*)&wpack[(size_t)(frag0 + nt * KTN + kt) * 512 + lane * 16];
      acc = __builtin_amdgcn_wmma_f32_16x16x32_f16(false, A[kt], false, Bf,
                                                   (short)0, acc, false, false);
    }
    if (FINAL) {
#pragma unroll
      for (int r = 0; r < 8; ++r)
        sOut[(wave * 16 + g * 8 + r) * OSTRIDE + n] = acc[r];
    } else {
      // scatter C/D layout -> next layer's A-fragment layout
      int kt2 = n >> 5, kl = n & 31;
      int v2 = ((kl >> 4) << 2) | ((kl >> 1) & 3);
      int gg = (kl >> 3) & 1, hf = kl & 1;
      int basecol = ((wave * 4 + kt2) * 32 + (gg << 4)) * 16 + (v2 << 1) + hf;
#pragma unroll
      for (int r = 0; r < 8; ++r) {
        float val = acc[r];
        val = val > 0.f ? val : 0.f;
        sAct[basecol + (g * 8 + r) * 16] = (_Float16)val;
      }
    }
  }
}

// ---------------------------------------------------------------------------
// Main fused kernel: one block = one ray (8 waves x 16 samples = 128 depths)
// ---------------------------------------------------------------------------
__global__ __launch_bounds__(256) void render_kernel(
    const float* __restrict__ rays_o, const float* __restrict__ rays_d,
    const float* __restrict__ grid,   const float* __restrict__ act_shift,
    const float* __restrict__ b0, const float* __restrict__ b1,
    const float* __restrict__ b2, const float* __restrict__ b3,
    const _Float16* __restrict__ wpack, const _Float16* __restrict__ lat,
    float* __restrict__ out) {
  __shared__ _Float16 sAct[WAVES * 4 * 32 * 16];  // 32 KB activation frags
  __shared__ float    sOut[SDEPTH * OSTRIDE];     // ~16.5 KB final layer out
  __shared__ float    sT[2][SDEPTH];              // scan ping-pong
  __shared__ float    sRGB[3 * SDEPTH];

  int ray = blockIdx.x, tid = threadIdx.x;
  int wave = tid >> 5, lane = tid & 31;

  float ox = rays_o[ray * 3 + 0], oy = rays_o[ray * 3 + 1], oz = rays_o[ray * 3 + 2];
  float dx = rays_d[ray * 3 + 0], dy = rays_d[ray * 3 + 1], dz = rays_d[ray * 3 + 2];

  // SH deg-2 basis (per-ray constant, kept in registers by every thread)
  float inv = rsqrtf(dx * dx + dy * dy + dz * dz);
  float nx = dx * inv, ny = dy * inv, nz = dz * inv;
  float bas[9];
  bas[0] = 0.28209479177387814f;
  bas[1] = -0.4886025119029199f * ny;
  bas[2] = 0.4886025119029199f * nz;
  bas[3] = -0.4886025119029199f * nx;
  bas[4] = 1.0925484305920792f * nx * ny;
  bas[5] = -1.0925484305920792f * ny * nz;
  bas[6] = 0.31539156525252005f * (2.f * nz * nz - nx * nx - ny * ny);
  bas[7] = -1.0925484305920792f * nx * nz;
  bas[8] = 0.5462742152960396f * (nx * nx - ny * ny);

  // ---- Phase 1: trilerp features + latent into A-fragment LDS (wave-local) --
  // Corner bases are vi*112 bytes -> always 16B-aligned: gather as float4.
  // even thread: channels 0..15 (4 float4 chunks); odd: 16..27 (3 chunks).
  {
    int m = tid >> 1;              // sample owned by this thread pair
    int wv = m >> 4, ml = m & 15;
    float t = ((float)m + 0.5f) * (1.0f / 128.0f);
    float px = ox + dx * t, py = oy + dy * t, pz = oz + dz * t;
    float u = fminf(fmaxf((px + 1.f) * 0.5f, 0.f), 1.f);
    float v = fminf(fmaxf((py + 1.f) * 0.5f, 0.f), 1.f);
    float w = fminf(fmaxf(pz, 0.f), 1.f);
    float posx = u * 255.f, posy = v * 255.f, posz = w * 127.f;
    int x0 = (int)posx; if (x0 > 254) x0 = 254;
    int y0 = (int)posy; if (y0 > 254) y0 = 254;
    int z0 = (int)posz; if (z0 > 126) z0 = 126;
    float fx = posx - (float)x0, fy = posy - (float)y0, fz = posz - (float)z0;
    float gx = 1.f - fx, gy = 1.f - fy, gz = 1.f - fz;
    float w000 = gx * gy * gz, w100 = fx * gy * gz, w010 = gx * fy * gz, w110 = fx * fy * gz;
    float w001 = gx * gy * fz, w101 = fx * gy * fz, w011 = gx * fy * fz, w111 = fx * fy * fz;
    const int CSX = 256 * 128 * 7, CSY = 128 * 7, CSZ = 7;   // float4 strides
    int base4 = ((x0 * 256 + y0) * 128 + z0) * 7;            // float4 index
    const float4* g4 = (const float4*)grid;
    int cb4  = (tid & 1) ? 4 : 0;      // chunk offset
    int nch  = (tid & 1) ? 3 : 4;      // chunks handled
    for (int ch = 0; ch < nch; ++ch) {
      int q = base4 + cb4 + ch;
      float4 f000 = g4[q];
      float4 f100 = g4[q + CSX];
      float4 f010 = g4[q + CSY];
      float4 f110 = g4[q + CSX + CSY];
      float4 f001 = g4[q + CSZ];
      float4 f101 = g4[q + CSX + CSZ];
      float4 f011 = g4[q + CSY + CSZ];
      float4 f111 = g4[q + CSX + CSY + CSZ];
      float vx = w000 * f000.x + w100 * f100.x + w010 * f010.x + w110 * f110.x +
                 w001 * f001.x + w101 * f101.x + w011 * f011.x + w111 * f111.x;
      float vy = w000 * f000.y + w100 * f100.y + w010 * f010.y + w110 * f110.y +
                 w001 * f001.y + w101 * f101.y + w011 * f011.y + w111 * f111.y;
      float vz = w000 * f000.z + w100 * f100.z + w010 * f010.z + w110 * f110.z +
                 w001 * f001.z + w101 * f101.z + w011 * f011.z + w111 * f111.z;
      float vw = w000 * f000.w + w100 * f100.w + w010 * f010.w + w110 * f110.w +
                 w001 * f001.w + w101 * f101.w + w011 * f011.w + w111 * f111.w;
      int cg = (cb4 + ch) * 4;
      *(v2h*)&sAct[frag_off(wv, ml, cg)]     = (v2h){(_Float16)vx, (_Float16)vy};
      *(v2h*)&sAct[frag_off(wv, ml, cg + 2)] = (v2h){(_Float16)vz, (_Float16)vw};
    }
    // latent channels 28..91 (packed pairs; 28 is even so pairs stay aligned)
    int j0 = (tid & 1) * 32;
    const v2h* lat2 = (const v2h*)lat;
    for (int jj = 0; jj < 32; jj += 2)
      *(v2h*)&sAct[frag_off(wv, ml, HASH_CH + j0 + jj)] = lat2[(j0 + jj) >> 1];
    if (tid & 1) {  // K-pad 92..95 for layer 0
      *(v2h*)&sAct[frag_off(wv, ml, 92)] = (v2h){(_Float16)0.f, (_Float16)0.f};
      *(v2h*)&sAct[frag_off(wv, ml, 94)] = (v2h){(_Float16)0.f, (_Float16)0.f};
    }
  }

  // ---- Phase 2: fused WMMA MLP (producers == consumers per wave; DS in-order)
  mlp_layer<3, 8, false>(sAct, sOut, wpack, wave, lane, 0,  b0, HID);
  mlp_layer<4, 8, false>(sAct, sOut, wpack, wave, lane, 24, b1, HID);
  mlp_layer<4, 8, false>(sAct, sOut, wpack, wave, lane, 56, b2, HID);
  mlp_layer<4, 2, true >(sAct, sOut, wpack, wave, lane, 88, b3, OUT_F);

  __syncthreads();

  // ---- Phase 3: alpha, transmittance scan, SH shading, marching ------------
  float alpha = 0.f;
  if (tid < SDEPTH) {
    float x = sOut[tid * OSTRIDE + 0] + act_shift[tid];
    float sp = (x > 20.f) ? x : log1pf(expf(x));   // softplus
    alpha = 1.f - expf(-sp * 2.0f);                 // interval = 256/128 = 2
    sT[0][tid] = 1.f - alpha + 1e-10f;              // q[s]
  }
  int src = 0;
  for (int off = 1; off < SDEPTH; off <<= 1) {      // inclusive product scan
    __syncthreads();
    if (tid < SDEPTH) {
      float vs = sT[src][tid];
      if (tid >= off) vs *= sT[src][tid - off];
      sT[1 - src][tid] = vs;
    }
    src ^= 1;
  }
  __syncthreads();
  if (tid < SDEPTH) {
    float T = (tid == 0) ? 1.f : sT[src][tid - 1];  // exclusive
    float wgt = alpha * T;
    for (int c = 0; c < 3; ++c) {
      float acc = 0.f;
      for (int k = 0; k < 9; ++k)
        acc += bas[k] * sOut[tid * OSTRIDE + 1 + c * 9 + k];
      float rgb = 1.f / (1.f + expf(-acc));
      sRGB[c * SDEPTH + tid] = wgt * rgb;
    }
  }
  __syncthreads();
  for (int st = 64; st > 0; st >>= 1) {
    if (tid < st) {
      sRGB[tid]              += sRGB[tid + st];
      sRGB[SDEPTH + tid]     += sRGB[SDEPTH + tid + st];
      sRGB[2 * SDEPTH + tid] += sRGB[2 * SDEPTH + tid + st];
    }
    __syncthreads();
  }
  if (tid == 0) {
    out[ray * 3 + 0] = sRGB[0];
    out[ray * 3 + 1] = sRGB[SDEPTH];
    out[ray * 3 + 2] = sRGB[2 * SDEPTH];
  }
}

// ---------------------------------------------------------------------------
extern "C" void kernel_launch(void* const* d_in, const int* in_sizes, int n_in,
                              void* d_out, int out_size, void* d_ws, size_t ws_size,
                              hipStream_t stream) {
  const float* rays_o  = (const float*)d_in[0];
  const float* rays_d  = (const float*)d_in[1];
  const float* times_t = (const float*)d_in[2];
  const float* grid    = (const float*)d_in[3];
  const float* ashift  = (const float*)d_in[4];
  const float* lat_w1  = (const float*)d_in[5];
  const float* lat_b1  = (const float*)d_in[6];
  const float* lat_w2  = (const float*)d_in[7];
  const float* lat_b2  = (const float*)d_in[8];
  const float* lat_w3  = (const float*)d_in[9];
  const float* lat_b3  = (const float*)d_in[10];
  const float* mlp_w0  = (const float*)d_in[11];
  const float* mlp_b0  = (const float*)d_in[12];
  const float* mlp_w1  = (const float*)d_in[13];
  const float* mlp_b1  = (const float*)d_in[14];
  const float* mlp_w2  = (const float*)d_in[15];
  const float* mlp_b2  = (const float*)d_in[16];
  const float* mlp_w3  = (const float*)d_in[17];
  const float* mlp_b3  = (const float*)d_in[18];
  float* outp = (float*)d_out;

  _Float16* wpack = (_Float16*)d_ws;          // 96 frags * 512 halves = 96 KB
  _Float16* latws = wpack + 96 * 512;         // 64 halves

  int n_rays = in_sizes[0] / 3;

  latent_kernel<<<1, 128, 0, stream>>>(times_t, lat_w1, lat_b1, lat_w2, lat_b2,
                                       lat_w3, lat_b3, latws);
  pack_kernel<<<96, 32, 0, stream>>>(mlp_w0, mlp_w1, mlp_w2, mlp_w3, wpack);
  render_kernel<<<n_rays, 256, 0, stream>>>(rays_o, rays_d, grid, ashift,
                                            mlp_b0, mlp_b1, mlp_b2, mlp_b3,
                                            wpack, latws, outp);
}
